// NaryLSTMLayer_50354196578418
// MI455X (gfx1250) — matrix operand order repaired and checked by reading, not compiled, not run on hardware
//
#include <hip/hip_runtime.h>

// Tree-LSTM (binary N-ary LSTM) for MI455X / gfx1250.
// L=24 sequential levels; per level: two GEMMs (x@W^T [256->512], hc@U^T [256->640])
// done with v_wmma_f32_16x16x32_bf16 (f32 accumulate), gather of child h/c, gates.
// Weights are repacked once to bf16 in a B-fragment-native [tile][col][K] layout.

#define L_LV 24
#define NN   8192
#define DIN  256
#define DD   128

typedef __attribute__((ext_vector_type(16))) __bf16 v16bf;
typedef __attribute__((ext_vector_type(8)))  float  v8f;

union Frag {
    uint4 q[2];   // 32 bytes = 8 VGPRs
    v16bf v;
};

__device__ __forceinline__ unsigned short f2bf(float f) {
    unsigned int u = __float_as_uint(f);
    u += 0x7fffu + ((u >> 16) & 1u);          // round-to-nearest-even
    return (unsigned short)(u >> 16);
}

__device__ __forceinline__ float sigmoidf_(float x) {
    return 1.0f / (1.0f + __expf(-x));
}

// ---------------------------------------------------------------------------
// One-shot: repack W_w [512,256] and stacked [U_f1;U_f2;U_iuo] [640,256] (fp32,
// row-major) into bf16 B-tile layout: p[ct][col][k], ct = 16-wide output-column
// tile, k contiguous per column so a lane's B fragment is two 16B reads.
// ---------------------------------------------------------------------------
__global__ void pack_weights_kernel(const float* __restrict__ Ww,
                                    const float* __restrict__ Uf1,
                                    const float* __restrict__ Uf2,
                                    const float* __restrict__ Uiuo,
                                    unsigned short* __restrict__ pW,
                                    unsigned short* __restrict__ pU) {
    const int WTOT = 32 * 16 * 256;   // 512 cols
    const int UTOT = 40 * 16 * 256;   // 640 cols
    int t = blockIdx.x * 256 + threadIdx.x;
    if (t < WTOT) {
        int k = t & 255, col = (t >> 8) & 15, ct = t >> 12;
        int j = ct * 16 + col;                     // output column = W_w row
        pW[t] = f2bf(Ww[j * 256 + k]);
    } else if (t < WTOT + UTOT) {
        int u = t - WTOT;
        int k = u & 255, col = (u >> 8) & 15, ct = u >> 12;
        int j = ct * 16 + col;                     // stacked U row
        float val;
        if (j < 128)        val = Uf1[j * 256 + k];
        else if (j < 256)   val = Uf2[(j - 128) * 256 + k];
        else                val = Uiuo[(j - 256) * 256 + k];
        pU[u] = f2bf(val);
    }
}

// ---------------------------------------------------------------------------
// One level. Block = 128 threads (4 waves) owns 16 nodes.
// LDS row stride 264 bf16 (528B = 132 dwords) -> lane m starts at bank 4m:
// conflict-free b128 A-fragment loads.
// ---------------------------------------------------------------------------
#define XSTRIDE 264

__global__ __launch_bounds__(128)
void level_kernel(const float* __restrict__ x,        // tensor + l*N*DIN
                  const int*   __restrict__ idx,      // indices + l*N*2
                  const float* __restrict__ h_init,   // [128]
                  const float* __restrict__ c_init,   // [128]
                  const float* __restrict__ Wb,       // [512]
                  const unsigned short* __restrict__ pW,   // [32][16][256] bf16
                  const unsigned short* __restrict__ pU,   // [40][16][256] bf16
                  const float* __restrict__ h_prev,   // level l-1 hidden
                  const float* __restrict__ c_prev,   // level l-1 cell
                  float* __restrict__ h_out,
                  float* __restrict__ c_out,
                  int first) {
    __shared__ __align__(16) unsigned short aX[16 * XSTRIDE];  // x tile, bf16
    __shared__ __align__(16) unsigned short aH[16 * XSTRIDE];  // hc tile, bf16

    const int tid   = threadIdx.x;
    const int node0 = blockIdx.x * 16;

    // ---- stage x[16,256] -> bf16 LDS (coalesced) ----
    #pragma unroll
    for (int it = 0; it < 32; ++it) {
        int item = tid + it * 128;
        int m = item >> 8, col = item & 255;
        aX[m * XSTRIDE + col] = f2bf(x[(node0 + m) * DIN + col]);
    }
    // ---- gather children hidden -> hc[16,256] -> bf16 LDS ----
    #pragma unroll
    for (int it = 0; it < 32; ++it) {
        int item = tid + it * 128;
        int m = item >> 8, col = item & 255;
        int child = col >> 7, d = col & 127;
        int ci = idx[(node0 + m) * 2 + child];
        float v = (first || ci < 0) ? h_init[d] : h_prev[(size_t)ci * DD + d];
        aH[m * XSTRIDE + col] = f2bf(v);
    }
    __syncthreads();

    const int lane = tid & 31;
    const int wave = tid >> 5;
    const int g    = lane >> 4;      // K-half / row-half group
    const int n    = lane & 15;      // B column / C column
    const unsigned short* rowX = &aX[(lane & 15) * XSTRIDE];
    const unsigned short* rowH = &aH[(lane & 15) * XSTRIDE];

    // Each wave owns two 16-wide d-slices: grp = wave, wave+4  (d = grp*16+n).
    // Output column tile for gate block j is ct = 8*j + grp for both GEMMs.
    for (int gi = 0; gi < 2; ++gi) {
        const int grp = wave + 4 * gi;

        v8f acc[9];
        #pragma unroll
        for (int t = 0; t < 9; ++t) acc[t] = v8f{};

        #pragma unroll
        for (int kt = 0; kt < 8; ++kt) {           // K = 256 in steps of 32
            // A fragments (shared by all 9 tiles this K-step)
            Frag fax, fah;
            const int e0 = kt * 32 + g * 8;        // element offset in row
            fax.q[0] = *(const uint4*)(rowX + e0);
            fax.q[1] = *(const uint4*)(rowX + e0 + 16);
            fah.q[0] = *(const uint4*)(rowH + e0);
            fah.q[1] = *(const uint4*)(rowH + e0 + 16);

            // W_x tiles: f,i,u,o  (cols 128*j + grp*16 + n)
            #pragma unroll
            for (int j = 0; j < 4; ++j) {
                const int ct = 8 * j + grp;
                const unsigned short* bp =
                    pW + ((size_t)(ct * 16 + n) * 256 + kt * 32 + g * 8);
                Frag b;
                b.q[0] = *(const uint4*)(bp);
                b.q[1] = *(const uint4*)(bp + 16);
                acc[j] = __builtin_amdgcn_wmma_f32_16x16x32_bf16(
                    false, fax.v, false, b.v, (short)0, acc[j], false, false);
            }
            // U tiles: f1,f2,i,u,o
            #pragma unroll
            for (int j = 0; j < 5; ++j) {
                const int ct = 8 * j + grp;
                const unsigned short* bp =
                    pU + ((size_t)(ct * 16 + n) * 256 + kt * 32 + g * 8);
                Frag b;
                b.q[0] = *(const uint4*)(bp);
                b.q[1] = *(const uint4*)(bp + 16);
                acc[4 + j] = __builtin_amdgcn_wmma_f32_16x16x32_bf16(
                    false, fah.v, false, b.v, (short)0, acc[4 + j], false, false);
            }
        }

        // ---- gates directly in WMMA C layout (row m = v + 8g, col d) ----
        const int d  = grp * 16 + n;
        const float bf = Wb[d];
        const float bi = Wb[128 + d];
        const float bu = Wb[256 + d];
        const float bo = Wb[384 + d];
        const float ci_d = c_init[d];

        #pragma unroll
        for (int v = 0; v < 8; ++v) {
            const int m    = v + 8 * g;
            const int node = node0 + m;
            const int i0 = idx[node * 2 + 0];
            const int i1 = idx[node * 2 + 1];
            float c0 = (first || i0 < 0) ? ci_d : c_prev[(size_t)i0 * DD + d];
            float c1 = (first || i1 < 0) ? ci_d : c_prev[(size_t)i1 * DD + d];

            float fx = acc[0][v] + bf;
            float ix = acc[1][v] + bi;
            float ux = acc[2][v] + bu;
            float ox = acc[3][v] + bo;

            float f1 = sigmoidf_(fx + acc[4][v]);
            float f2 = sigmoidf_(fx + acc[5][v]);
            float ig = sigmoidf_(ix + acc[6][v]);
            float ug = tanhf(ux + acc[7][v]);
            float og = sigmoidf_(ox + acc[8][v]);

            float nc = ig * ug + f1 * c0 + f2 * c1;
            float nh = og * tanhf(nc);
            h_out[(size_t)node * DD + d] = nh;
            c_out[(size_t)node * DD + d] = nc;
        }
    }
}

// ---------------------------------------------------------------------------
extern "C" void kernel_launch(void* const* d_in, const int* in_sizes, int n_in,
                              void* d_out, int out_size, void* d_ws, size_t ws_size,
                              hipStream_t stream) {
    (void)in_sizes; (void)n_in; (void)out_size; (void)ws_size;
    const float* tensor  = (const float*)d_in[0];
    const int*   indices = (const int*)  d_in[1];
    const float* h_init  = (const float*)d_in[2];
    const float* c_init  = (const float*)d_in[3];
    const float* W_w     = (const float*)d_in[4];
    const float* W_b     = (const float*)d_in[5];
    const float* U_f1    = (const float*)d_in[6];
    const float* U_f2    = (const float*)d_in[7];
    const float* U_iuo   = (const float*)d_in[8];

    float* h_all = (float*)d_out;                          // [L,N,D]
    float* c_all = h_all + (size_t)L_LV * NN * DD;         // [L,N,D]

    unsigned short* pW = (unsigned short*)d_ws;            // 32*16*256 bf16
    unsigned short* pU = pW + 32 * 16 * 256;               // 40*16*256 bf16

    // repack weights to bf16 B-fragment layout (every call; deterministic)
    pack_weights_kernel<<<(32 * 16 * 256 + 40 * 16 * 256) / 256, 256, 0, stream>>>(
        W_w, U_f1, U_f2, U_iuo, pW, pU);

    // 24 strictly sequential levels
    for (int l = 0; l < L_LV; ++l) {
        const float* hp = l ? (h_all + (size_t)(l - 1) * NN * DD) : h_init;
        const float* cp = l ? (c_all + (size_t)(l - 1) * NN * DD) : c_init;
        level_kernel<<<NN / 16, 128, 0, stream>>>(
            tensor + (size_t)l * NN * DIN,
            indices + (size_t)l * NN * 2,
            h_init, c_init, W_b, pW, pU,
            hp, cp,
            h_all + (size_t)l * NN * DD,
            c_all + (size_t)l * NN * DD,
            l == 0 ? 1 : 0);
    }
}
